// Encoder_2250562863755
// MI455X (gfx1250) — matrix operand order repaired and checked by reading, not compiled, not run on hardware
//
#include <hip/hip_runtime.h>

// Problem constants (match reference)
#define BB 128
#define SS 512
#define EE 1024
#define HH 1024
#define G4 4096          // 4*H  (gate rows: i,f,g,o)
#define KK 2048          // E + H (fused K)

typedef __bf16 bf16_t;
typedef __attribute__((ext_vector_type(16))) __bf16 v16bf;
typedef __attribute__((ext_vector_type(8)))  __bf16 v8bf;
typedef __attribute__((ext_vector_type(8)))  float  v8f;

// ---- fast activations (clamped so exp never produces inf/NaN path) ----
__device__ __forceinline__ float fast_sigmoid(float x) {
    x = fminf(fmaxf(x, -30.0f), 30.0f);
    return 1.0f / (1.0f + __expf(-x));
}
__device__ __forceinline__ float fast_tanh(float x) {
    x = fminf(fmaxf(x, -15.0f), 15.0f);
    float e = __expf(-2.0f * x);
    return (1.0f - e) / (1.0f + e);
}

// A-fragment (16x32 bf16, M x K): lane L (M=L&15) holds K = {(L>>4)*8 .. +7} and
// {16+(L>>4)*8 .. +7} relative to the K-block base. `p` is already lane-adjusted:
// row_base + kb*32 + (lane>>4)*8. Two 16-byte loads.
__device__ __forceinline__ v16bf load_fragA(const bf16_t* p) {
    v8bf lo = *(const v8bf*)(p);
    v8bf hi = *(const v8bf*)(p + 16);
    v16bf r;
#pragma unroll
    for (int i = 0; i < 8; ++i) { r[i] = lo[i]; r[i + 8] = hi[i]; }
    return r;
}

// Wpack layout: [wv(32)][kb(64)][nt(8)][lane(32)][e(16)] bf16.
// For wave wv, K-block kb, tile nt=g*2+s: lane (=half*16+l15) holds the 16
// K elements {kb*32 + half*8 .. +7} and {kb*32 + 16 + half*8 .. +7} of weight row
// r = g*H + wv*32 + s*16 + l15 (B-fragment register image, fetch-ordered).
#define WV_STRIDE (64 * 8 * 32 * 16)   // elements per wave block (256 K)

// ---------------------------------------------------------------------------
// Fused 1-layer LSTM scan. Grid = B/16 blocks (independent batch tiles),
// 1024 threads = 32 wave32 per block (8 waves/SIMD for latency hiding).
// Wave w owns hidden columns [32w, 32w+32): 2 WMMA N-tiles per gate x 4 gates
// = 8 f32 accumulators; c-state lives in registers for the whole sequence.
// h is exchanged via double-buffered LDS, one barrier per time step.
// ---------------------------------------------------------------------------
template <int WRITE_F32>
__global__ __launch_bounds__(1024, 1)
void lstm_layer_kernel(const bf16_t* __restrict__ xin,    // [B,S,E] bf16
                       const bf16_t* __restrict__ Wpack,  // swizzled weights, see above
                       const float*  __restrict__ bias,   // [4H] = b_ih + b_hh
                       bf16_t* __restrict__ seq_bf16,     // layer0: [B,S,H] bf16 out
                       float*  __restrict__ seq_f32,      // layer1: [B,S,H] f32 out
                       float*  __restrict__ h_final,      // [B,H] f32
                       float*  __restrict__ c_final)      // [B,H] f32
{
    __shared__ bf16_t hbuf[2][16][HH + 8];   // +8 bf16 pad: row stride 516 dw -> bank-spread

    const int tid   = threadIdx.x;
    const int lane  = tid & 31;
    const int wv    = tid >> 5;       // wave 0..31
    const int half  = lane >> 4;      // 0/1
    const int l15   = lane & 15;
    const int mrow0 = half * 8;       // first of 8 C-fragment rows held by this lane
    const int bt    = blockIdx.x;     // batch tile (16 rows)

    // h_{-1} = 0
    {
        bf16_t* z = &hbuf[0][0][0];
        for (int i = tid; i < 16 * (HH + 8); i += 1024) z[i] = (bf16_t)0.0f;
    }
    __syncthreads();

    // Per-lane global A base (x part): row = batch row, K-contiguous layout.
    const bf16_t* xbase = xin + ((size_t)(bt * 16 + l15) * SS) * EE + half * 8;

    // Wave-uniform weight base; per-lane part is just lane*16 elements (32 B).
    // All 8 fragments of K-block kb sit at byte offsets (kb*8+nt)*1024 (< 512 KB),
    // so every load is saddr + small immediate: no per-iteration address VALU.
    const bf16_t* wvb  = Wpack + (size_t)wv * WV_STRIDE;
    const int     loff = lane * 16;

    float bia[8];
#pragma unroll
    for (int g = 0; g < 4; ++g)
#pragma unroll
        for (int s = 0; s < 2; ++s)
            bia[g * 2 + s] = bias[g * HH + wv * 32 + s * 16 + l15];

    v8f c_acc[2];
#pragma unroll
    for (int s = 0; s < 2; ++s)
#pragma unroll
        for (int k = 0; k < 8; ++k) c_acc[s][k] = 0.0f;

    int p = 0;
#pragma unroll 1
    for (int t = 0; t < SS; ++t) {
        v8f acc[8];
#pragma unroll
        for (int nt = 0; nt < 8; ++nt)
#pragma unroll
            for (int k = 0; k < 8; ++k) acc[nt][k] = bia[nt];

        // ---- input half: gates += x_t @ W_ih^T  (K blocks 0..31, A from global) ----
        const bf16_t* xt = xbase + (size_t)t * EE;
#pragma unroll 2
        for (int kb = 0; kb < 32; ++kb) {
            v16bf a = load_fragA(xt + kb * 32);
            v16bf bf[8];
#pragma unroll
            for (int nt = 0; nt < 8; ++nt)
                bf[nt] = *(const v16bf*)(wvb + (size_t)(kb * 8 + nt) * 512 + loff);
#pragma unroll
            for (int nt = 0; nt < 8; ++nt)
                acc[nt] = __builtin_amdgcn_wmma_f32_16x16x32_bf16(
                    false, a, false, bf[nt], (short)0, acc[nt], false, false);
        }

        // ---- recurrent half: gates += h_{t-1} @ W_hh^T (K blocks 32..63, A from LDS) ----
        const bf16_t* hp = &hbuf[p][l15][half * 8];
#pragma unroll 2
        for (int kb = 0; kb < 32; ++kb) {
            v16bf a = load_fragA(hp + kb * 32);
            v16bf bf[8];
#pragma unroll
            for (int nt = 0; nt < 8; ++nt)
                bf[nt] = *(const v16bf*)(wvb + (size_t)((kb + 32) * 8 + nt) * 512 + loff);
#pragma unroll
            for (int nt = 0; nt < 8; ++nt)
                acc[nt] = __builtin_amdgcn_wmma_f32_16x16x32_bf16(
                    false, a, false, bf[nt], (short)0, acc[nt], false, false);
        }

        // ---- gate nonlinearity + cell update (i/f/g/o tiles are lane-aligned) ----
        const int np = p ^ 1;
#pragma unroll
        for (int s = 0; s < 2; ++s) {
            v8f iv = acc[0 * 2 + s];
            v8f fv = acc[1 * 2 + s];
            v8f gv = acc[2 * 2 + s];
            v8f ov = acc[3 * 2 + s];
            const int j = wv * 32 + s * 16 + l15;   // hidden column
#pragma unroll
            for (int k = 0; k < 8; ++k) {
                float ig = fast_sigmoid(iv[k]);
                float fg = fast_sigmoid(fv[k]);
                float gg = fast_tanh(gv[k]);
                float og = fast_sigmoid(ov[k]);
                float c  = fg * c_acc[s][k] + ig * gg;
                c_acc[s][k] = c;
                float h  = og * fast_tanh(c);

                hbuf[np][mrow0 + k][j] = (bf16_t)h;

                const int brow = bt * 16 + mrow0 + k;
                const size_t oidx = ((size_t)brow * SS + t) * HH + j;
                if (WRITE_F32) seq_f32[oidx] = h;
                else           seq_bf16[oidx] = (bf16_t)h;
                if (t == SS - 1) h_final[(size_t)brow * HH + j] = h;
            }
        }
        __syncthreads();
        p = np;
    }

    // final cell state
#pragma unroll
    for (int s = 0; s < 2; ++s)
#pragma unroll
        for (int k = 0; k < 8; ++k) {
            const int brow = bt * 16 + mrow0 + k;
            c_final[(size_t)brow * HH + wv * 32 + s * 16 + l15] = c_acc[s][k];
        }
}

// ---------------------------------------------------------------------------
// Prep kernels.
// ---------------------------------------------------------------------------
__global__ void embed_bf16_kernel(const int* __restrict__ src,
                                  const float* __restrict__ emb,
                                  bf16_t* __restrict__ xout) {
    const int bs = blockIdx.x;                 // b*S + s, 0..65535
    const int row = src[bs];
    const float* e = emb + (size_t)row * EE;
    bf16_t* o = xout + (size_t)bs * EE;
    for (int i = threadIdx.x; i < EE; i += blockDim.x) o[i] = (bf16_t)e[i];
}

// Swizzle fp32 W_ih/W_hh into the fetch-ordered bf16 Wpack layout
// [wv][kb][nt][lane][e]  (8,388,608 elements total).
__global__ void pack_w_kernel(const float* __restrict__ Wih,
                              const float* __restrict__ Whh,
                              bf16_t* __restrict__ Wpack) {
    const size_t idx = (size_t)blockIdx.x * 1024 + threadIdx.x;
    const int e    = (int)(idx & 15);
    const int lane = (int)((idx >> 4) & 31);
    const int nt   = (int)((idx >> 9) & 7);
    const int kb   = (int)((idx >> 12) & 63);
    const int wv   = (int)(idx >> 18);
    const int g = nt >> 1, s = nt & 1;
    const int half = lane >> 4, l15 = lane & 15;
    const int r = g * HH + wv * 32 + s * 16 + l15;   // weight row (gate output)
    const int c = kb * 32 + ((e >> 3) << 4) + half * 8 + (e & 7);   // fused K column
    const float v = (c < EE) ? Wih[(size_t)r * EE + c]
                             : Whh[(size_t)r * HH + (c - EE)];
    Wpack[idx] = (bf16_t)v;
}

__global__ void bias_kernel(const float* __restrict__ bih,
                            const float* __restrict__ bhh,
                            float* __restrict__ bsum) {
    const int i = blockIdx.x * blockDim.x + threadIdx.x;
    if (i < G4) bsum[i] = bih[i] + bhh[i];
}

// ---------------------------------------------------------------------------
extern "C" void kernel_launch(void* const* d_in, const int* in_sizes, int n_in,
                              void* d_out, int out_size, void* d_ws, size_t ws_size,
                              hipStream_t stream) {
    (void)in_sizes; (void)n_in; (void)out_size; (void)ws_size;

    const int*   src  = (const int*)  d_in[0];
    const float* emb  = (const float*)d_in[1];
    const float* Wih0 = (const float*)d_in[2];
    const float* Whh0 = (const float*)d_in[3];
    const float* bih0 = (const float*)d_in[4];
    const float* bhh0 = (const float*)d_in[5];
    const float* Wih1 = (const float*)d_in[6];
    const float* Whh1 = (const float*)d_in[7];
    const float* bih1 = (const float*)d_in[8];
    const float* bhh1 = (const float*)d_in[9];
    float* out = (float*)d_out;

    // workspace layout
    char* ws = (char*)d_ws;
    size_t off = 0;
    bf16_t* x0   = (bf16_t*)(ws + off); off += (size_t)BB * SS * EE * 2;   // 128 MB
    bf16_t* out0 = (bf16_t*)(ws + off); off += (size_t)BB * SS * HH * 2;   // 128 MB
    bf16_t* Wp0  = (bf16_t*)(ws + off); off += (size_t)G4 * KK * 2;        // 16 MB
    bf16_t* Wp1  = (bf16_t*)(ws + off); off += (size_t)G4 * KK * 2;        // 16 MB
    float*  bs0  = (float*)(ws + off);  off += (size_t)G4 * 4;
    float*  bs1  = (float*)(ws + off);  off += (size_t)G4 * 4;

    const size_t SEQ = (size_t)BB * SS * HH;   // out1 elements
    float* hsec = out + SEQ;                   // h: [2,B,H]
    float* csec = out + SEQ + 2 * (size_t)BB * HH;  // c: [2,B,H]

    embed_bf16_kernel<<<BB * SS, 256, 0, stream>>>(src, emb, x0);
    pack_w_kernel<<<(G4 * KK) / 1024, 1024, 0, stream>>>(Wih0, Whh0, Wp0);
    pack_w_kernel<<<(G4 * KK) / 1024, 1024, 0, stream>>>(Wih1, Whh1, Wp1);
    bias_kernel<<<G4 / 256, 256, 0, stream>>>(bih0, bhh0, bs0);
    bias_kernel<<<G4 / 256, 256, 0, stream>>>(bih1, bhh1, bs1);

    // layer 0: seq output -> bf16 scratch (feeds layer 1)
    lstm_layer_kernel<0><<<BB / 16, 1024, 0, stream>>>(
        x0, Wp0, bs0, out0, nullptr, hsec, csec);
    // layer 1: seq output -> fp32 d_out
    lstm_layer_kernel<1><<<BB / 16, 1024, 0, stream>>>(
        out0, Wp1, bs1, nullptr, out,
        hsec + (size_t)BB * HH, csec + (size_t)BB * HH);
}
